// AutoregressiveConvLSTM_34840774705339
// MI455X (gfx1250) — matrix operand order repaired and checked by reading, not compiled, not run on hardware
//
#include <hip/hip_runtime.h>

typedef __attribute__((ext_vector_type(16))) _Float16 v16h;
typedef __attribute__((ext_vector_type(8)))  float    v8f;

#define Bn 64
#define Cn 16
#define Hd 128
#define Wd 128
#define OWN 16
#define HALO 17
#define ROWS (OWN + 2 * HALO)        // 50 stored rows per strip
#define NPIX (ROWS * Wd)             // 6400
#define NTILES (NPIX / 16)           // 400 tiles of 16 px (tiles never span rows: 128/16=8)
#define NTHREADS 512
#define NWAVES (NTHREADS / 32)       // 16
#define TPW (NTILES / NWAVES)        // 25 tiles per wave
#define STRIPS (Hd / OWN)            // 8
#define HALF_LOG_2PI 0.9189385332046727f

// LDS layout (float offsets)
#define H0_OFS  0
#define H1_OFS  (2 * NPIX)
#define CF_OFS  (4 * NPIX)
#define SCR_OFS (6 * NPIX)
#define CST_OFS (8 * NPIX)
#define GST_OFS (10 * NPIX)
#define ZP_OFS  (10 * NPIX + NWAVES * 128)   // single zero-pad word
#define SMEM_FLOATS (ZP_OFS + 1)

__device__ __forceinline__ float sigmoidf(float x) { return 1.0f / (1.0f + __expf(-x)); }

__device__ __forceinline__ v8f wmma16(v16h a, v16h b, v8f c) {
  // D = A(16x32 f16) * B(32x16 f16) + C(16x16 f32)
  return __builtin_amdgcn_wmma_f32_16x16x32_f16(false, a, false, b, (short)0, c, false, false);
}

// Feature index K (shared A/B decode, pure bit ops, no division):
//   dcol = K & 3, drow = (K >> 2) & 3, ci = K >> 4
//   valid tap iff dcol < 3 && drow < 3 (and ci < #channels)

// Build a B fragment (32x16 f16): lane L holds N = L&15, element e holds K = e + 8*(e>=8) + 8*(L>>4).
// Weight tensor is HWIO: W[(tap*IC + CBASE + ci)*OC + n], valid for ci < NC, n < OC.
// Branchless: clamp index, load unconditionally, mask the value (load + v_cndmask, no exec branch).
template <int IC, int CBASE, int OC, int NC>
__device__ __forceinline__ v16h buildB(const float* __restrict__ W, int g, int n16) {
  v16h out;
#pragma unroll
  for (int e = 0; e < 16; ++e) {
    int K = e + ((e >= 8) ? 8 : 0) + 8 * g;
    int dcol = K & 3, drow = (K >> 2) & 3, ci = K >> 4;
    bool tv = (dcol < 3) && (drow < 3) && (ci < NC) && (n16 < OC);
    int tap = drow * 3 + dcol;
    int idx = tv ? ((tap * IC + CBASE + ci) * OC + n16) : 0;  // always-safe address
    float wv = W[idx];
    out[e] = (_Float16)(tv ? wv : 0.f);
  }
  return out;
}

// Branchless A-fragment gather from a strip-local LDS field.
// A layout (16-bit, 16x32): lane L covers M = L&15, element e holds K = e + 8*(e>=8) + 8*(L>>4).
// Out-of-bounds taps read the zero-pad word via address select (no exec branches).
template <bool TWO_CH>
__device__ __forceinline__ v16h gatherA(const float* __restrict__ sm, int fofs, int tile,
                                        int g8 /* 8*(lane>>4) */, int m /* lane&15 */,
                                        int rlo, int rhi) {
  const int px = tile * 16 + m;
  const int r = px >> 7, c = px & (Wd - 1);
  float fv[16];
#pragma unroll
  for (int e = 0; e < 16; ++e) {
    const int ep = e + ((e >= 8) ? 8 : 0);
    int k = ep + g8;
    int dcol = k & 3;
    int drow = (k >> 2) & 3;
    int rr = r + drow - 1;
    int cc = c + dcol - 1;
    bool ok = (dcol < 3) && (drow < 3) &&
              ((unsigned)(rr - rlo) < (unsigned)(rhi - rlo)) &&
              ((unsigned)cc < (unsigned)Wd);
    if (!TWO_CH) ok = ok && (k < 16);                  // 1-channel field: ci must be 0
    int chofs = TWO_CH ? ((k & 16) * (NPIX / 16)) : 0; // ci*NPIX
    int ii = fofs + chofs + px + (drow - 1) * Wd + (dcol - 1);
    fv[e] = sm[ok ? ii : ZP_OFS];
  }
  v16h a;
#pragma unroll
  for (int e = 0; e < 16; ++e) a[e] = (_Float16)fv[e];
  return a;
}

__global__ void __launch_bounds__(NTHREADS, 4)   // 16 waves resident = 4/SIMD -> 256 VGPR budget
convlstm_strip_kernel(const float* __restrict__ xg,  const float* __restrict__ cg,
                      const float* __restrict__ Wci, const float* __restrict__ bci,
                      const float* __restrict__ Wc1, const float* __restrict__ bc1,
                      const float* __restrict__ Wc2, const float* __restrict__ bc2,
                      const float* __restrict__ Wo1, const float* __restrict__ bo1,
                      const float* __restrict__ Wo2, const float* __restrict__ bo2,
                      const float* __restrict__ Wih, const float* __restrict__ bih,
                      const float* __restrict__ Whh, float* __restrict__ ws) {
  extern __shared__ float smem[];

  const int tid   = threadIdx.x;
  const int lane  = tid & 31;
  const int wv    = tid >> 5;
  const int g     = lane >> 4;
  const int g8    = g << 3;
  const int n16   = lane & 15;
  const int b     = blockIdx.x >> 3;
  const int strip = blockIdx.x & (STRIPS - 1);
  const int grow0 = strip * OWN - HALO;            // global row of local row 0
  const int rlo   = (grow0 < 0) ? -grow0 : 0;      // first locally-stored in-image row
  const int rhiv  = (Hd - grow0 < ROWS) ? (Hd - grow0) : ROWS;
  float* wgst = smem + GST_OFS + wv * 128;

  float wci[9];
#pragma unroll
  for (int t9 = 0; t9 < 9; ++t9) wci[t9] = Wci[t9];
  const float bci0 = bci[0];
  const float bc10 = bc1[0], bc11 = bc1[1];
  // branchless bias loads: clamped index + value mask
  const float bihn = (n16 < 8) ? bih[n16 & 7] : 0.f;
  const float bo1n = (n16 < 2) ? bo1[n16 & 1] : 0.f;
  const float bo2n = (n16 < 2) ? bo2[n16 & 1] : 0.f;

  // ---- init h=0, c=0, zero-pad word
  for (int i = tid; i < 2 * NPIX; i += NTHREADS) { smem[H0_OFS + i] = 0.f; smem[CST_OFS + i] = 0.f; }
  if (tid == 0) smem[ZP_OFS] = 0.f;

  // ---- cond inner field: tanh(conv3x3(cond C=16 -> 2) + bc1); branchless border masking
  for (int px = tid; px < NPIX; px += NTHREADS) {
    int r = px >> 7, c = px & (Wd - 1);
    int gr = grow0 + r;
    float a0 = bc10, a1 = bc11;
    const float* cp = cg + ((size_t)b * Cn) * Hd * Wd;
    if (gr >= 0 && gr < Hd) {   // whole-lane validity (uniform rows of work)
#pragma unroll
      for (int tap = 0; tap < 9; ++tap) {
        int rr = gr + tap / 3 - 1, cc = c + tap % 3 - 1;
        bool ok = ((unsigned)rr < (unsigned)Hd) && ((unsigned)cc < (unsigned)Wd);
        int off = ok ? (rr * Wd + cc) : 0;               // clamped, always-safe address
#pragma unroll
        for (int ci = 0; ci < Cn; ++ci) {
          float v = cp[(size_t)ci * Hd * Wd + off];
          float w0 = ok ? Wc1[(tap * 16 + ci) * 2 + 0] : 0.f;
          float w1 = ok ? Wc1[(tap * 16 + ci) * 2 + 1] : 0.f;
          a0 += v * w0;
          a1 += v * w1;
        }
      }
    }
    smem[SCR_OFS + px] = tanhf(a0);
    smem[SCR_OFS + NPIX + px] = tanhf(a1);
  }
  __syncthreads();

  // ---- cond_f = conv3x3(inner, Wc2) + bc2, via WMMA (Bc2 frag scoped to this phase)
  {
    const v16h Bc2 = buildB<2, 0, 2, 2>(Wc2, g, n16);
    const float bc2n = (n16 < 2) ? bc2[n16 & 1] : 0.f;
    for (int j = 0; j < TPW; ++j) {
      int tile = wv * TPW + j;
      v8f d = {};
      d = wmma16(gatherA<true>(smem, SCR_OFS, tile, g8, n16, rlo, rhiv), Bc2, d);
      if (n16 < 2) {
#pragma unroll
        for (int rr = 0; rr < 8; ++rr)
          smem[CF_OFS + n16 * NPIX + tile * 16 + (rr + 8 * g)] = d[rr] + bc2n;
      }
    }
  }
  __syncthreads();

  // ---- main-loop B fragments (5 live: 40 VGPRs, within the 256-VGPR budget)
  const v16h Bgu = buildB<1, 0, 8, 1>(Wih, g, n16);   // (3,3,1,8)
  const v16h Bgh = buildB<2, 0, 8, 2>(Whh, g, n16);   // (3,3,2,8)
  const v16h B1a = buildB<4, 0, 2, 2>(Wo1, g, n16);   // (3,3,4,2) inputs 0..1 = h
  const v16h B1b = buildB<4, 2, 2, 2>(Wo1, g, n16);   // (3,3,4,2) inputs 2..3 = cond_f
  const v16h B2  = buildB<2, 0, 2, 2>(Wo2, g, n16);   // (3,3,2,2)

  // ---- channel recurrence
  float lp = 0.f;
  int hcur = H0_OFS, hnxt = H1_OFS;
  for (int t = 0; t < Cn; ++t) {
    if (t > 0) {
      // u = conv3x3(x[ch t-1], Wci) + bci  (x from global; working set is L2-resident)
      const float* xp = xg + (size_t)(b * Cn + (t - 1)) * Hd * Wd;
      for (int px = tid; px < NPIX; px += NTHREADS) {
        int r = px >> 7, c = px & (Wd - 1);
        int gr = grow0 + r;
        float acc = bci0;
#pragma unroll
        for (int tap = 0; tap < 9; ++tap) {
          int rr = gr + tap / 3 - 1, cc = c + tap % 3 - 1;
          bool ok = ((unsigned)rr < (unsigned)Hd) && ((unsigned)cc < (unsigned)Wd);
          int off = ok ? (rr * Wd + cc) : 0;
          float w = ok ? wci[tap] : 0.f;
          acc += w * xp[off];
        }
        smem[SCR_OFS + px] = acc;
        if ((unsigned)gr < (unsigned)Hd)  // warm path for channel t (used in log-prob)
          __builtin_prefetch(xp + (size_t)Hd * Wd + gr * Wd + c, 0, 0);
      }
      __syncthreads();

      // gates = conv(u,Wih) + conv(h,Whh) + bih; LSTM pointwise update
      for (int j = 0; j < TPW; ++j) {
        int tile = wv * TPW + j;
        v8f d = {};
        d = wmma16(gatherA<false>(smem, SCR_OFS, tile, g8, n16, rlo, rhiv), Bgu, d);
        d = wmma16(gatherA<true >(smem, hcur,    tile, g8, n16, rlo, rhiv), Bgh, d);
        if (n16 < 8) {
#pragma unroll
          for (int rr = 0; rr < 8; ++rr)
            wgst[(rr + 8 * g) * 8 + n16] = d[rr] + bihn;   // D[M][N] -> stage[M][gate]
        }
        asm volatile("s_wait_dscnt 0x0" ::: "memory");
        {
          int m = n16, f = g;                 // lane handles (pixel m, channel f)
          int px = tile * 16 + m;
          float gi = wgst[m * 8 + 0 + f];
          float gf = wgst[m * 8 + 2 + f];
          float gg = wgst[m * 8 + 4 + f];
          float go = wgst[m * 8 + 6 + f];
          float cp = smem[CST_OFS + f * NPIX + px];
          float cn = sigmoidf(gf) * cp + sigmoidf(gi) * tanhf(gg);
          smem[CST_OFS + f * NPIX + px] = cn;
          smem[hnxt + f * NPIX + px] = sigmoidf(go) * tanhf(cn);
        }
      }
      __syncthreads();
      { int tmp = hcur; hcur = hnxt; hnxt = tmp; }
    }

    // t-field = relu(conv3x3(concat(h, cond_f), Wo1) + bo1), two K-chunks of WMMA
    for (int j = 0; j < TPW; ++j) {
      int tile = wv * TPW + j;
      v8f d = {};
      d = wmma16(gatherA<true>(smem, hcur,   tile, g8, n16, rlo, rhiv), B1a, d);
      d = wmma16(gatherA<true>(smem, CF_OFS, tile, g8, n16, rlo, rhiv), B1b, d);
      if (n16 < 2) {
#pragma unroll
        for (int rr = 0; rr < 8; ++rr)
          smem[SCR_OFS + n16 * NPIX + tile * 16 + (rr + 8 * g)] = fmaxf(d[rr] + bo1n, 0.f);
      }
    }
    __syncthreads();

    // params = conv3x3(t-field, Wo2) + bo2; accumulate Gaussian log-prob on own rows
    const float* xt = xg + (size_t)(b * Cn + t) * Hd * Wd;
    for (int j = 0; j < TPW; ++j) {
      int tile = wv * TPW + j;
      v8f d = {};
      d = wmma16(gatherA<true>(smem, SCR_OFS, tile, g8, n16, rlo, rhiv), B2, d);
      if (n16 < 2) {
#pragma unroll
        for (int rr = 0; rr < 8; ++rr)
          wgst[(rr + 8 * g) * 2 + n16] = d[rr] + bo2n;     // stage[M][{mean,log_std}]
      }
      asm volatile("s_wait_dscnt 0x0" ::: "memory");
      if (lane < 16) {
        int px = tile * 16 + lane;
        int r = px >> 7, c = px & (Wd - 1);
        if (r >= HALO && r < HALO + OWN) {                  // own (always-valid) rows only
          int gr = grow0 + r;
          float mean = wgst[lane * 2 + 0];
          float ls   = wgst[lane * 2 + 1];
          float xv   = xt[gr * Wd + c];
          float z = (xv - mean) * __expf(-ls);
          lp += -0.5f * z * z - ls - HALF_LOG_2PI;
        }
      }
    }
    __syncthreads();
  }

  // ---- deterministic reduction: wave32 shuffle tree, then cross-wave via LDS
#pragma unroll
  for (int o = 16; o > 0; o >>= 1) lp += __shfl_down(lp, o, 32);
  if (lane == 0) smem[GST_OFS + wv] = lp;
  __syncthreads();
  if (tid == 0) {
    float s = 0.f;
#pragma unroll
    for (int i = 0; i < NWAVES; ++i) s += smem[GST_OFS + i];
    ws[blockIdx.x] = s;
  }
}

__global__ void reduce_strips_kernel(const float* __restrict__ ws, float* __restrict__ out) {
  int b = threadIdx.x;
  if (b < Bn) {
    float s = 0.f;
#pragma unroll
    for (int i = 0; i < STRIPS; ++i) s += ws[b * STRIPS + i];
    out[b] = s;
  }
}

extern "C" void kernel_launch(void* const* d_in, const int* in_sizes, int n_in,
                              void* d_out, int out_size, void* d_ws, size_t ws_size,
                              hipStream_t stream) {
  (void)in_sizes; (void)n_in; (void)out_size; (void)ws_size;
  const float* x   = (const float*)d_in[0];
  const float* cnd = (const float*)d_in[1];
  const float* Wci = (const float*)d_in[2];
  const float* bci = (const float*)d_in[3];
  const float* Wc1 = (const float*)d_in[4];
  const float* bc1 = (const float*)d_in[5];
  const float* Wc2 = (const float*)d_in[6];
  const float* bc2 = (const float*)d_in[7];
  const float* Wo1 = (const float*)d_in[8];
  const float* bo1 = (const float*)d_in[9];
  const float* Wo2 = (const float*)d_in[10];
  const float* bo2 = (const float*)d_in[11];
  const float* Wih = (const float*)d_in[12];
  const float* bih = (const float*)d_in[13];
  const float* Whh = (const float*)d_in[14];
  float* ws = (float*)d_ws;

  // ~258 KB of the 320 KB WGP LDS
  size_t shmem = (size_t)SMEM_FLOATS * sizeof(float);
  convlstm_strip_kernel<<<dim3(Bn * STRIPS), dim3(NTHREADS), shmem, stream>>>(
      x, cnd, Wci, bci, Wc1, bc1, Wc2, bc2, Wo1, bo1, Wo2, bo2, Wih, bih, Whh, ws);
  reduce_strips_kernel<<<dim3(1), dim3(64), 0, stream>>>(ws, (float*)d_out);
}
